// CConvEncoder_34041910788792
// MI455X (gfx1250) — compile-verified, matrix operand order checked by reading, not run on hardware
//
#include <hip/hip_runtime.h>
#include <math.h>

// ---------------- problem constants ----------------
#define BATCH   2
#define NPTS    4096
#define CIN     128
#define COUT    256
#define NG      32
#define MGRID   32768          // NG^3
#define KNN     16
#define NTAP    27
#define KD      (NTAP*CIN)     // 3456 = reduction dim of main GEMM
#define KSTEPS  (KD/32)        // 108 wmma k-steps
#define NTILES  (COUT/16)      // 16 output column tiles
#define TILE_M  16
#define CHUNK   512
#define DXF     0.03125f
#define RADF    0.078125f      // 2.5 * DX
#define R2F     (RADF*RADF)
#define EPSF    1e-12f

typedef __bf16 bf16_t;
typedef __attribute__((ext_vector_type(16))) __bf16 v16bf;
typedef __attribute__((ext_vector_type(8)))  __bf16 v8bf;
typedef __attribute__((ext_vector_type(8)))  float  v8f;

#define WFS 3464   // 3456 + 8 pad -> row stride 1732 dwords == 4 mod 64 banks

struct Smem {
    bf16_t wf[TILE_M][WFS];                       // 110848 B, bf16 A-matrix rows
    union {
        struct {                                  // phase 1 scratch
            float candd[TILE_M][16][KNN];
            int   candi[TILE_M][16][KNN];
            float chunk[CHUNK*3];
        } p1;
        struct {                                  // phase 2/3 scratch
            float wlds[TILE_M][KNN][NTAP];
        } p2;
    } u;
    int   idxl[TILE_M][KNN];
    float invc[TILE_M];
};

__device__ __forceinline__ float fsign(float v){
    return (v > 0.f) ? 1.f : ((v < 0.f) ? -1.f : 0.f);
}

__device__ __forceinline__ void insert16(float d, int i, float* bd, int* bi){
    bd[KNN-1] = d; bi[KNN-1] = i;
#pragma unroll
    for (int q = KNN-1; q > 0; --q){
        if (bd[q] < bd[q-1]){
            float td = bd[q]; bd[q] = bd[q-1]; bd[q-1] = td;
            int   ti = bi[q]; bi[q] = bi[q-1]; bi[q-1] = ti;
        }
    }
}

// Open3D ball_to_cube_volume_preserving (scalar port of the reference)
__device__ __forceinline__ void ball_to_cube(float x, float y, float z,
                                             float& qx, float& qy, float& qz){
    float rho2 = x*x + y*y;
    float r2   = rho2 + z*z;
    float r    = sqrtf(r2 + EPSF);
    float rho  = sqrtf(rho2 + EPSF);
    bool  cap  = (1.25f*z*z > rho2);
    float s_cap  = sqrtf(3.0f*r / (r + fabsf(z) + EPSF));
    float s_side = r / rho;
    float s  = cap ? s_cap : s_side;
    float cx = s*x, cy = s*y;
    float cz = cap ? (fsign(z)*r) : (1.5f*z);
    if (r2 < 1e-20f){ cx = 0.f; cy = 0.f; cz = 0.f; }
    float crho = sqrtf(cx*cx + cy*cy + EPSF);
    bool  xbig = fabsf(cx) > fabsf(cy);
    float major = xbig ? cx : cy;
    float minor = xbig ? cy : cx;
    float major_safe = (fabsf(major) < EPSF) ? 1.0f : major;
    float t = (4.0f/3.14159265358979323846f) * atanf(minor/major_safe);
    float uu = fsign(major)*crho;
    float vv = uu*t;
    float px = xbig ? uu : vv;
    float py = xbig ? vv : uu;
    if (cx*cx + cy*cy < 1e-20f){ px = 0.f; py = 0.f; }
    qx = px; qy = py; qz = cz;
}

// trilinear tap weights along one axis (align_corners, 3-tap grid)
__device__ __forceinline__ void axis_w(float c, float* w){
    float id = (c*0.5f + 0.5f)*2.0f;
    id = fminf(fmaxf(id, 0.0f), 2.0f);
    float i0 = fminf(fmaxf(floorf(id), 0.0f), 1.0f);
    float fr = fminf(fmaxf(id - i0, 0.0f), 1.0f);
    int ii = (int)i0;
    w[0] = 0.f; w[1] = 0.f; w[2] = 0.f;
    if (ii == 0){ w[0] = 1.0f - fr; w[1] = fr; }
    else        { w[1] = 1.0f - fr; w[2] = fr; }
}

// ---- prep: repack kernel [3456][256] f32 -> bf16 in WMMA B-fragment order ----
// layout: [nt][ks][lane][16]; lane<16 -> n=nt*16+lane, K j<8 : hi*8+j ; j>=8 : 16+hi*8+j-8
__global__ void pack_kernel_bf16(const float* __restrict__ kern, bf16_t* __restrict__ bpack){
    int flat = blockIdx.x*256 + threadIdx.x;
    const int total = NTILES*KSTEPS*32*16;
    if (flat >= total) return;
    int j    = flat & 15;
    int lane = (flat >> 4) & 31;
    int rest = flat >> 9;                 // nt*KSTEPS + ks
    int ks   = rest % KSTEPS;
    int nt   = rest / KSTEPS;
    int hi   = lane >> 4;
    int n    = nt*16 + (lane & 15);
    int kl   = (j < 8) ? (hi*8 + j) : (16 + hi*8 + (j - 8));
    int k    = ks*32 + kl;
    bpack[flat] = (bf16_t)kern[(size_t)k*COUT + n];
}

// ---- fused kernel: KNN + continuous-conv weights + wf + WMMA GEMM ----
__global__ __launch_bounds__(256)
void cconv_fused(const float* __restrict__ feat, const float* __restrict__ pos,
                 const bf16_t* __restrict__ bpack, float* __restrict__ out){
    extern __shared__ char smem_raw[];
    Smem* sm = (Smem*)smem_raw;

    const int b  = blockIdx.y;
    const int mt = blockIdx.x;              // grid tile: 16 rows
    const float* pos_b  = pos  + (size_t)b*NPTS*3;
    const float* feat_b = feat + (size_t)b*NPTS*CIN;

    // ---------------- phase 1: KNN (radius-filtered top-16) ----------------
    {
        const int g = threadIdx.x >> 4;     // grid row in tile
        const int s = threadIdx.x & 15;     // scan slice
        const int m = mt*TILE_M + g;
        const float gx = ((m >> 10)      + 0.5f)*DXF;
        const float gy = (((m >> 5)&31)  + 0.5f)*DXF;
        const float gz = ((m & 31)       + 0.5f)*DXF;

        float bd[KNN]; int bi[KNN];
#pragma unroll
        for (int q = 0; q < KNN; ++q){ bd[q] = 3.0e38f; bi[q] = -1; }

        for (int c = 0; c < NPTS; c += CHUNK){
            for (int uidx = threadIdx.x; uidx < CHUNK*3; uidx += 256)
                sm->u.p1.chunk[uidx] = pos_b[c*3 + uidx];
            __syncthreads();
            for (int j = s; j < CHUNK; j += 16){
                float dx = sm->u.p1.chunk[j*3+0] - gx;
                float dy = sm->u.p1.chunk[j*3+1] - gy;
                float dz = sm->u.p1.chunk[j*3+2] - gz;
                float d2 = dx*dx + dy*dy + dz*dz;
                if (d2 <= R2F && d2 < bd[KNN-1]) insert16(d2, c + j, bd, bi);
            }
            __syncthreads();
        }
#pragma unroll
        for (int q = 0; q < KNN; ++q){
            sm->u.p1.candd[g][s][q] = bd[q];
            sm->u.p1.candi[g][s][q] = bi[q];
        }
    }
    __syncthreads();

    // merge 16 partial lists per grid row (one thread per row)
    if (threadIdx.x < TILE_M){
        const int g = threadIdx.x;
        float md[KNN]; int mi[KNN];
#pragma unroll
        for (int q = 0; q < KNN; ++q){ md[q] = 3.0e38f; mi[q] = -1; }
        for (int s2 = 0; s2 < 16; ++s2)
            for (int q = 0; q < KNN; ++q){
                float d = sm->u.p1.candd[g][s2][q];
                int   i = sm->u.p1.candi[g][s2][q];
                if (i >= 0 && d < md[KNN-1]) insert16(d, i, md, mi);
            }
        int cnt = 0;
#pragma unroll
        for (int q = 0; q < KNN; ++q){ sm->idxl[g][q] = mi[q]; cnt += (mi[q] >= 0); }
        sm->invc[g] = 1.0f / (float)((cnt > 0) ? cnt : 1);
    }
    __syncthreads();

    // ---------------- phase 2: 27 tap weights per (row, neighbor) ----------------
    {
        const int g = threadIdx.x >> 4;
        const int k = threadIdx.x & 15;
        const int m = mt*TILE_M + g;
        const float gx = ((m >> 10)      + 0.5f)*DXF;
        const float gy = (((m >> 5)&31)  + 0.5f)*DXF;
        const float gz = ((m & 31)       + 0.5f)*DXF;
        const int gi = sm->idxl[g][k];
        if (gi >= 0){
            float px = pos_b[gi*3+0], py = pos_b[gi*3+1], pz = pos_b[gi*3+2];
            float qx, qy, qz;
            ball_to_cube((px-gx)/RADF, (py-gy)/RADF, (pz-gz)/RADF, qx, qy, qz);
            float wx[3], wy[3], wz[3];
            axis_w(qx, wx); axis_w(qy, wy); axis_w(qz, wz);
#pragma unroll
            for (int tz = 0; tz < 3; ++tz)
#pragma unroll
                for (int ty = 0; ty < 3; ++ty){
                    float wzy = wz[tz]*wy[ty];
#pragma unroll
                    for (int tx = 0; tx < 3; ++tx)
                        sm->u.p2.wlds[g][k][tz*9 + ty*3 + tx] = wzy*wx[tx];
                }
        } else {
#pragma unroll
            for (int t = 0; t < NTAP; ++t) sm->u.p2.wlds[g][k][t] = 0.0f;
        }
    }
    __syncthreads();

    // ---------------- phase 3: wf[g][t*128+i] = (1/cnt) * sum_k w * feat ----------------
    {
        const int g  = threadIdx.x >> 4;
        const int s  = threadIdx.x & 15;
        const int i0 = s*8;
        float f[KNN][8];
#pragma unroll
        for (int k = 0; k < KNN; ++k){
            int gi = sm->idxl[g][k];
            if (gi >= 0){
                const float* fr = feat_b + (size_t)gi*CIN + i0;
#pragma unroll
                for (int q = 0; q < 8; ++q) f[k][q] = fr[q];
            } else {
#pragma unroll
                for (int q = 0; q < 8; ++q) f[k][q] = 0.0f;
            }
        }
        const float ic = sm->invc[g];
        for (int t = 0; t < NTAP; ++t){
            float acc[8];
#pragma unroll
            for (int q = 0; q < 8; ++q) acc[q] = 0.0f;
#pragma unroll
            for (int k = 0; k < KNN; ++k){
                float w = sm->u.p2.wlds[g][k][t];
#pragma unroll
                for (int q = 0; q < 8; ++q) acc[q] += w*f[k][q];
            }
            v8bf vv;
#pragma unroll
            for (int q = 0; q < 8; ++q) vv[q] = (bf16_t)(acc[q]*ic);
            *(v8bf*)&sm->wf[g][t*CIN + i0] = vv;
        }
    }
    __syncthreads();

    // ---------------- phase 4: WMMA GEMM wf[16,3456] x kernel[3456,256] ----------------
    {
        const int lane = threadIdx.x & 31;
        const int wave = threadIdx.x >> 5;     // 8 waves, 2 col tiles each
        const int mrow = lane & 15;
        const int hi   = lane >> 4;
        const int nt0  = wave*2, nt1 = wave*2 + 1;
        const bf16_t* bp0 = bpack + ((size_t)nt0*KSTEPS*32 + lane)*16;
        const bf16_t* bp1 = bpack + ((size_t)nt1*KSTEPS*32 + lane)*16;
        v8f c0 = {}; v8f c1 = {};
        for (int ks = 0; ks < KSTEPS; ++ks){
            // A fragment (documented 16-bit A layout: lanes<16 K0-7/16-23, lanes>=16 K8-15/24-31)
            const v8bf* ap = (const v8bf*)&sm->wf[mrow][ks*32 + hi*8];
            v8bf alo = ap[0];
            v8bf ahi = ap[2];                  // +16 bf16 elements
            v16bf a;
#pragma unroll
            for (int q = 0; q < 8; ++q){ a[q] = alo[q]; a[q+8] = ahi[q]; }
            // B fragments, pre-packed contiguous per lane
            const v8bf* b0p = (const v8bf*)(bp0 + (size_t)ks*32*16);
            const v8bf* b1p = (const v8bf*)(bp1 + (size_t)ks*32*16);
            if (ks + 1 < KSTEPS){
                __builtin_prefetch(bp0 + (size_t)(ks+1)*32*16, 0, 3);
                __builtin_prefetch(bp1 + (size_t)(ks+1)*32*16, 0, 3);
            }
            v8bf b0l = b0p[0], b0h = b0p[1], b1l = b1p[0], b1h = b1p[1];
            v16bf b0, b1;
#pragma unroll
            for (int q = 0; q < 8; ++q){
                b0[q] = b0l[q]; b0[q+8] = b0h[q];
                b1[q] = b1l[q]; b1[q+8] = b1h[q];
            }
            c0 = __builtin_amdgcn_wmma_f32_16x16x32_bf16(false, a, false, b0,
                                                         (short)0, c0, false, false);
            c1 = __builtin_amdgcn_wmma_f32_16x16x32_bf16(false, a, false, b1,
                                                         (short)0, c1, false, false);
        }
        // C layout: VGPR r -> M = r + 8*hi, N = lane&15 ; out[b][o][m]
        const size_t ob = (size_t)b*COUT*MGRID;
        const int    mg = mt*TILE_M;
#pragma unroll
        for (int r = 0; r < 8; ++r){
            int ml = r + hi*8;
            out[ob + (size_t)(nt0*16 + mrow)*MGRID + mg + ml] = c0[r];
            out[ob + (size_t)(nt1*16 + mrow)*MGRID + mg + ml] = c1[r];
        }
    }
}

extern "C" void kernel_launch(void* const* d_in, const int* in_sizes, int n_in,
                              void* d_out, int out_size, void* d_ws, size_t ws_size,
                              hipStream_t stream) {
    const float* feat = (const float*)d_in[0];   // [2,4096,128]
    const float* pos  = (const float*)d_in[1];   // [2,4096,3]
    // d_in[2] = grid_pos (recomputed exactly on device: (i+0.5)/32)
    const float* kern = (const float*)d_in[3];   // [27*128*256] = [3456][256]
    float*   out   = (float*)d_out;              // [2,256,32,32,32]
    bf16_t*  bpack = (bf16_t*)d_ws;              // 1.73 MB packed bf16 B matrix

    const int packTotal = NTILES*KSTEPS*32*16;   // 884736
    pack_kernel_bf16<<<(packTotal + 255)/256, 256, 0, stream>>>(kern, bpack);

    const size_t smemBytes = sizeof(Smem);       // ~151 KB (<= 320 KB/WGP on CDNA5)
    cconv_fused<<<dim3(MGRID/TILE_M, BATCH), dim3(256), smemBytes, stream>>>(
        feat, pos, bpack, out);
}
